// RbfSplitNet_19842748908187
// MI455X (gfx1250) — compile-verified, matrix operand order checked by reading, not run on hardware
//
#include <hip/hip_runtime.h>
#include <hip/hip_bf16.h>
#include <math.h>

// ---------------------------------------------------------------------------
// RBF continuous-conv GNN, restructured as:
//   (1) per-edge basis coefficients (VALU)
//   (2) sorted-target scatter: A[n, k*Cin+i] += coeff[e,k]*f[src,i]  (no atomics)
//   (3) dense GEMM  out = A_bf16 @ Wpack_bf16  via v_wmma_f32_16x16x32_bf16,
//       one wave per 16-row strip computing ALL column tiles (A-fragment reuse)
//   (4) VALU epilogues (bias / fc / residual / /128)
// ---------------------------------------------------------------------------

typedef __attribute__((ext_vector_type(16))) __bf16 bfvec16;
typedef __attribute__((ext_vector_type(8)))  float  floatx8;

struct __attribute__((aligned(16))) U128 { unsigned int x, y, z, w; };
union BfPack { bfvec16 v; U128 q[2]; };

__device__ __forceinline__ unsigned short f2bf(float f) {
  unsigned int u = __float_as_uint(f);
  u += 0x7FFFu + ((u >> 16) & 1u);          // round to nearest even
  return (unsigned short)(u >> 16);
}

__device__ __forceinline__ int lowerBound(const int* __restrict__ a, int n, int key) {
  int lo = 0, hi = n;
  while (lo < hi) {
    int mid = (lo + hi) >> 1;
    if (a[mid] < key) lo = mid + 1; else hi = mid;
  }
  return lo;
}

// ---------------------------------------------------------------------------
// (1) per-edge hat-basis coefficients.  d = sgn*(posS[src]-posT[tgt])/SUPPORT
// fluid:  f_d = (posT - posS)/S  -> sgn = -1
// bound:  b_d = (posS - posT)/S  -> sgn = +1
// ---------------------------------------------------------------------------
__global__ void coef_kernel(const float* __restrict__ posT, const float* __restrict__ posS,
                            const int* __restrict__ tgt, const int* __restrict__ src,
                            float sgn, int E, float* __restrict__ coef) {
  int e = blockIdx.x * blockDim.x + threadIdx.x;
  if (e >= E) return;
  int ti = tgt[e], si = src[e];
  const float invS = 1.0f / 1.5f;
  float dx = sgn * (posS[si * 2 + 0] - posT[ti * 2 + 0]) * invS;
  float dy = sgn * (posS[si * 2 + 1] - posT[ti * 2 + 1]) * invS;
  dx = fminf(fmaxf(dx, -1.f), 1.f);
  dy = fminf(fmaxf(dy, -1.f), 1.f);
  float r  = sqrtf(dx * dx + dy * dy) * 2.f - 1.f;
  float th = atan2f(dy, dx) * 0.318309886183790672f;   // 1/pi
  float hr[4], ht[4];
#pragma unroll
  for (int m = 0; m < 4; ++m) {
    float c = -1.f + (2.f / 3.f) * (float)m;           // centers of 4 hats
    hr[m] = fmaxf(0.f, 1.f - fabsf(r  - c) * 1.5f);    // 1/spacing = 1.5
    ht[m] = fmaxf(0.f, 1.f - fabsf(th - c) * 1.5f);
  }
  float* cp = coef + (size_t)e * 16;
#pragma unroll
  for (int a = 0; a < 4; ++a)
#pragma unroll
    for (int b = 0; b < 4; ++b)
      cp[a * 4 + b] = hr[a] * ht[b];
}

// ---------------------------------------------------------------------------
// (2) scatter:  one wave per target node (tgt is sorted).  A row accumulated
// in registers (lane owns idx = lane + 32j), written as bf16 [NF][KPAD].
// ---------------------------------------------------------------------------
template <int CIN, bool RELU>
__global__ void scatter_kernel(const float* __restrict__ F,
                               const int* __restrict__ src,
                               const int* __restrict__ tgt,
                               const float* __restrict__ coef,
                               int E, int Nnodes,
                               unsigned short* __restrict__ Abf) {
  constexpr int KI    = 16 * CIN;
  constexpr int NACC  = (KI + 31) / 32;
  constexpr int KPAD  = NACC * 32;
  constexpr int FREGS = (CIN + 31) / 32;

  const int lane = threadIdx.x & 31;
  const int n = blockIdx.x * (blockDim.x >> 5) + (threadIdx.x >> 5);
  if (n >= Nnodes) return;

  const int lo  = lowerBound(tgt, E, n);
  const int hiE = lowerBound(tgt, E, n + 1);

  float acc[NACC];
#pragma unroll
  for (int j = 0; j < NACC; ++j) acc[j] = 0.f;

  for (int e = lo; e < hiE; ++e) {
    int s = src[e];
    float cf = (lane < 16) ? coef[(size_t)e * 16 + lane] : 0.f;
    float fv[FREGS];
#pragma unroll
    for (int t = 0; t < FREGS; ++t) {
      int idx = lane + 32 * t;
      float x = (idx < CIN) ? F[(size_t)s * CIN + idx] : 0.f;
      if (RELU) x = fmaxf(x, 0.f);
      fv[t] = x;
    }
#pragma unroll
    for (int j = 0; j < NACC; ++j) {
      int idx = lane + 32 * j;
      if (idx < KI) {
        int k = idx / CIN;
        int i = idx - k * CIN;
        float c = __shfl(cf, k, 32);
        // shuffle each fixed register, then select per-lane (correct shfl use)
        float fx = __shfl(fv[0], i & 31, 32);
        if (FREGS > 1) { float f1 = __shfl(fv[1], i & 31, 32); if ((i >> 5) == 1) fx = f1; }
        if (FREGS > 2) { float f2 = __shfl(fv[2], i & 31, 32); if ((i >> 5) == 2) fx = f2; }
        acc[j] += c * fx;
      }
    }
  }
  unsigned short* arow = Abf + (size_t)n * KPAD;
#pragma unroll
  for (int j = 0; j < NACC; ++j) arow[lane + 32 * j] = f2bf(acc[j]);
}

// ---------------------------------------------------------------------------
// (3a) repack concatenated weight pair [Wa|Wb] (f32 [K][CoHalf] each) into the
// bf16 WMMA B-operand lane layout: Wpack[(nt*KB+kb)*32 + lane][16]
//   element e <-> K = kb*32 + (e/8)*16 + (lane>=16 ? 8:0) + (e%8), N = nt*16 + (lane&15)
// ---------------------------------------------------------------------------
__global__ void repack_w_kernel(const float* __restrict__ Wa, const float* __restrict__ Wb,
                                unsigned short* __restrict__ Wp,
                                int Kreal, int Kpad, int CoHalf) {
  int t = blockIdx.x * blockDim.x + threadIdx.x;
  int KB = Kpad >> 5;
  int ntiles = (CoHalf * 2) >> 4;
  int total = ntiles * KB * 32;
  if (t >= total) return;
  int lane = t & 31;
  int kb = (t >> 5) % KB;
  int nt = (t >> 5) / KB;
  int N = (nt << 4) + (lane & 15);
  int hi = (lane >> 4) << 3;                 // 0 or 8
  unsigned short* dst = Wp + (size_t)t * 16;
#pragma unroll
  for (int e = 0; e < 16; ++e) {
    int K = (kb << 5) + ((e >> 3) << 4) + hi + (e & 7);
    float v = 0.f;
    if (K < Kreal)
      v = (N < CoHalf) ? Wa[(size_t)K * CoHalf + N]
                       : Wb[(size_t)K * CoHalf + (N - CoHalf)];
    dst[e] = f2bf(v);
  }
}

// ---------------------------------------------------------------------------
// (3b) WMMA GEMM: out[M][Co] = A_bf16[M][Kpad] @ Wpack.
// One wave per 16-row strip; the A fragment is loaded once per K-step and
// reused for all NT=Co/16 column tiles (NT back-to-back WMMAs per K-step).
// ---------------------------------------------------------------------------
template <int NT>
__global__ void wmma_gemm_kernel(const unsigned short* __restrict__ Abf,
                                 const unsigned short* __restrict__ Wp,
                                 float* __restrict__ out,
                                 int M, int Kpad) {
  const int Co = NT * 16;
  const int lane = threadIdx.x & 31;
  const int wave = threadIdx.x >> 5;
  const int mt = blockIdx.x * 8 + wave;
  if (mt >= (M >> 4)) return;
  const int KB = Kpad >> 5;
  const int hi = (lane >> 4) << 3;           // 0 or 8
  const int row = (mt << 4) + (lane & 15);

  const unsigned short* arow  = Abf + (size_t)row * Kpad + hi;
  const unsigned short* wlane = Wp + (size_t)lane * 16;

  floatx8 acc[NT];
#pragma unroll
  for (int nt = 0; nt < NT; ++nt)
    acc[nt] = (floatx8){0.f, 0.f, 0.f, 0.f, 0.f, 0.f, 0.f, 0.f};

  for (int kb = 0; kb < KB; ++kb) {
    BfPack a;
    a.q[0] = *(const U128*)(arow + kb * 32);        // K = kb*32 + hi + [0..7]
    a.q[1] = *(const U128*)(arow + kb * 32 + 16);   // K = kb*32 + hi + [16..23]
#pragma unroll
    for (int nt = 0; nt < NT; ++nt) {
      BfPack b;
      const unsigned short* wp = wlane + (size_t)(nt * KB + kb) * 512;
      b.q[0] = *(const U128*)(wp);
      b.q[1] = *(const U128*)(wp + 8);
      acc[nt] = __builtin_amdgcn_wmma_f32_16x16x32_bf16(false, a.v, false, b.v,
                                                        (short)0, acc[nt], false, false);
    }
  }
  const int colBase = lane & 15;
#pragma unroll
  for (int nt = 0; nt < NT; ++nt)
#pragma unroll
    for (int v = 0; v < 8; ++v)
      out[(size_t)((mt << 4) + hi + v) * Co + (nt << 4) + colBase] = acc[nt][v];
}

// ---------------------------------------------------------------------------
// (4) epilogues
// ---------------------------------------------------------------------------
__global__ void l1_combine_kernel(const float* __restrict__ fluidFeats,
                                  const float* __restrict__ fcW0, const float* __restrict__ fcb0,
                                  const float* __restrict__ convF, const float* __restrict__ convB,
                                  const float* __restrict__ b0, const float* __restrict__ b1,
                                  const float* __restrict__ b2, const float* __restrict__ b3,
                                  float* __restrict__ ans1, int NFn) {
  int t = blockIdx.x * blockDim.x + threadIdx.x;
  if (t >= NFn * 96) return;
  int n = t / 96, c = t - n * 96;
  float v;
  if (c < 32) {
    v = fluidFeats[n * 2 + 0] * fcW0[c] + fluidFeats[n * 2 + 1] * fcW0[32 + c] + fcb0[c];
  } else if (c < 64) {
    int o = c - 32;
    v = convF[(size_t)n * 32 + o] + (o < 16 ? b0[o] : b1[o - 16]);
  } else {
    int o = c - 64;
    v = convB[(size_t)n * 32 + o] + (o < 16 ? b2[o] : b3[o - 16]);
  }
  ans1[t] = v;
}

__global__ void l2_combine_kernel(const float* __restrict__ conv,
                                  const float* __restrict__ b4, const float* __restrict__ b5,
                                  const float* __restrict__ ans1,
                                  const float* __restrict__ fcW1, const float* __restrict__ fcb1,
                                  float* __restrict__ ans2, int NFn) {
  int t = blockIdx.x * blockDim.x + threadIdx.x;
  if (t >= NFn * 64) return;
  int n = t / 64, o = t - n * 64;
  float fc = fcb1[o];
  const float* arow = ans1 + (size_t)n * 96;
  for (int i = 0; i < 96; ++i) fc += fmaxf(arow[i], 0.f) * fcW1[i * 64 + o];
  ans2[t] = conv[t] + (o < 32 ? b4[o] : b5[o - 32]) + fc;
}

__global__ void l3_combine_kernel(const float* __restrict__ conv,
                                  const float* __restrict__ b6, const float* __restrict__ b7,
                                  const float* __restrict__ ans2,
                                  const float* __restrict__ fcW2, const float* __restrict__ fcb2,
                                  float* __restrict__ out, int NFn) {
  int t = blockIdx.x * blockDim.x + threadIdx.x;
  if (t >= NFn * 64) return;
  int n = t / 64, o = t - n * 64;
  float fc = fcb2[o];
  const float* arow = ans2 + (size_t)n * 64;
  for (int i = 0; i < 64; ++i) fc += fmaxf(arow[i], 0.f) * fcW2[i * 64 + o];
  out[t] = (conv[t] + (o < 32 ? b6[o] : b7[o - 32]) + fc + ans2[t]) * (1.f / 128.f);
}

// ---------------------------------------------------------------------------
extern "C" void kernel_launch(void* const* d_in, const int* in_sizes, int n_in,
                              void* d_out, int out_size, void* d_ws, size_t ws_size,
                              hipStream_t stream) {
  const float* fluidPos   = (const float*)d_in[0];
  const float* boundPos   = (const float*)d_in[1];
  const float* fluidFeats = (const float*)d_in[2];
  const float* boundFeats = (const float*)d_in[3];
  const int*   fe_i = (const int*)d_in[4];   // sorted -> tgt
  const int*   fe_j = (const int*)d_in[5];   // src
  const int*   be_f = (const int*)d_in[6];   // sorted -> tgt
  const int*   be_b = (const int*)d_in[7];   // src
  const float* W0 = (const float*)d_in[8];   const float* b0 = (const float*)d_in[9];
  const float* W1 = (const float*)d_in[10];  const float* b1 = (const float*)d_in[11];
  const float* W2 = (const float*)d_in[12];  const float* b2 = (const float*)d_in[13];
  const float* W3 = (const float*)d_in[14];  const float* b3 = (const float*)d_in[15];
  const float* W4 = (const float*)d_in[16];  const float* b4 = (const float*)d_in[17];
  const float* W5 = (const float*)d_in[18];  const float* b5 = (const float*)d_in[19];
  const float* W6 = (const float*)d_in[20];  const float* b6 = (const float*)d_in[21];
  const float* W7 = (const float*)d_in[22];  const float* b7 = (const float*)d_in[23];
  const float* fcW0 = (const float*)d_in[24]; const float* fcb0 = (const float*)d_in[25];
  const float* fcW1 = (const float*)d_in[26]; const float* fcb1 = (const float*)d_in[27];
  const float* fcW2 = (const float*)d_in[28]; const float* fcb2 = (const float*)d_in[29];

  const int NFn = in_sizes[0] / 2;
  const int EF  = in_sizes[4];
  const int EB  = in_sizes[6];

  char* ws = (char*)d_ws;
  size_t off = 0;
  auto alloc = [&](size_t bytes) -> char* {
    char* p = ws + off;
    off += (bytes + 255) & ~(size_t)255;
    return p;
  };
  float*          coefF = (float*)alloc((size_t)EF * 16 * 4);
  float*          coefB = (float*)alloc((size_t)EB * 16 * 4);
  unsigned short* Abig  = (unsigned short*)alloc((size_t)NFn * 1536 * 2);
  unsigned short* Af1   = (unsigned short*)alloc((size_t)NFn * 32 * 2);
  unsigned short* Ab1   = (unsigned short*)alloc((size_t)NFn * 64 * 2);
  unsigned short* Wp    = (unsigned short*)alloc((size_t)64 * 1536 * 2);
  float*          convF = (float*)alloc((size_t)NFn * 32 * 4);
  float*          convB = (float*)alloc((size_t)NFn * 32 * 4);
  float*          conv2 = (float*)alloc((size_t)NFn * 64 * 4);
  float*          ans1  = (float*)alloc((size_t)NFn * 96 * 4);
  float*          ans2  = (float*)alloc((size_t)NFn * 64 * 4);

  // (1) edge coefficients
  coef_kernel<<<(EF + 255) / 256, 256, 0, stream>>>(fluidPos, fluidPos, fe_i, fe_j, -1.f, EF, coefF);
  coef_kernel<<<(EB + 255) / 256, 256, 0, stream>>>(fluidPos, boundPos, be_f, be_b, +1.f, EB, coefB);

  const int scatterBlocks = (NFn + 7) / 8;   // 8 waves/block, one node/wave
  const int gemmBlocks = ((NFn >> 4) + 7) / 8;

  auto repack = [&](const float* Wa, const float* Wb, int Kreal, int Kpad, int CoHalf) {
    int total = (((CoHalf * 2) >> 4) * (Kpad >> 5)) * 32;
    repack_w_kernel<<<(total + 255) / 256, 256, 0, stream>>>(Wa, Wb, Wp, Kreal, Kpad, CoHalf);
  };

  // layer 1
  scatter_kernel<2, false><<<scatterBlocks, 256, 0, stream>>>(fluidFeats, fe_j, fe_i, coefF, EF, NFn, Af1);
  scatter_kernel<3, false><<<scatterBlocks, 256, 0, stream>>>(boundFeats, be_b, be_f, coefB, EB, NFn, Ab1);
  repack(W0, W1, 32, 32, 16);
  wmma_gemm_kernel<2><<<gemmBlocks, 256, 0, stream>>>(Af1, Wp, convF, NFn, 32);
  repack(W2, W3, 48, 64, 16);
  wmma_gemm_kernel<2><<<gemmBlocks, 256, 0, stream>>>(Ab1, Wp, convB, NFn, 64);
  l1_combine_kernel<<<(NFn * 96 + 255) / 256, 256, 0, stream>>>(
      fluidFeats, fcW0, fcb0, convF, convB, b0, b1, b2, b3, ans1, NFn);

  // layer 2: convs 4,5 share A (Cin=96) + fc1
  scatter_kernel<96, true><<<scatterBlocks, 256, 0, stream>>>(ans1, fe_j, fe_i, coefF, EF, NFn, Abig);
  repack(W4, W5, 1536, 1536, 32);
  wmma_gemm_kernel<4><<<gemmBlocks, 256, 0, stream>>>(Abig, Wp, conv2, NFn, 1536);
  l2_combine_kernel<<<(NFn * 64 + 255) / 256, 256, 0, stream>>>(
      conv2, b4, b5, ans1, fcW1, fcb1, ans2, NFn);

  // layer 3: convs 6,7 share A (Cin=64) + fc2 + residual, /128
  scatter_kernel<64, true><<<scatterBlocks, 256, 0, stream>>>(ans2, fe_j, fe_i, coefF, EF, NFn, Abig);
  repack(W6, W7, 1024, 1024, 32);
  wmma_gemm_kernel<4><<<gemmBlocks, 256, 0, stream>>>(Abig, Wp, conv2, NFn, 1024);
  l3_combine_kernel<<<(NFn * 64 + 255) / 256, 256, 0, stream>>>(
      conv2, b6, b7, ans2, fcW2, fcb2, (float*)d_out, NFn);
}
